// DSConv_37726992728315
// MI455X (gfx1250) — matrix operand order repaired and checked by reading, not compiled, not run on hardware
//
#include <hip/hip_runtime.h>
#include <hip/hip_bf16.h>

typedef __attribute__((ext_vector_type(16))) _Float16 v16h;
typedef __attribute__((ext_vector_type(8)))  float    v8f;

// Problem constants: B=4, C=64, H=W=128, K=9, OUT=64, KK=K*C=576
// ws layout (bytes):
//   offws : 0        .. 2359296   (4*9*16384 f32)  per-pixel snake offsets
//   wpack : 2359296  .. 2433024   (576*64 f16)     B-fragment-ordered weights
//   gstats: 2433024  .. 2433536   (64 keys * 2 f32) groupnorm sum/sumsq
//   xws   : 2433536  .. 19210752  (4*64*16384 f32) pre-GN conv output (NCHW)

// ---------------------------------------------------------------------------
// Kernel P: pack conv_w (OUT,C,9,1) into WMMA B-fragment order, zero GN stats.
// Flat index = (((s*4+nt)*2 + khalf)*16 + n)*16 + e, kglobal = s*32+khalf*16+e,
// kglobal = ktap*64 + c  (K-dim order must match kMain's gather).
// ---------------------------------------------------------------------------
__global__ __launch_bounds__(256) void kPack(const float* __restrict__ cw,
                                             _Float16* __restrict__ wp,
                                             float* __restrict__ gstats)
{
    int idx = blockIdx.x * 256 + threadIdx.x;
    if (blockIdx.x == 0 && threadIdx.x < 128) gstats[threadIdx.x] = 0.0f;
    if (idx >= 36864) return;
    int e  = idx & 15;
    int n  = (idx >> 4) & 15;
    int kh = (idx >> 8) & 1;
    int nt = (idx >> 9) & 3;
    int s  = idx >> 11;
    int kg = s * 32 + kh * 16 + e;          // 0..575
    int kt = kg >> 6;                        // tap 0..8
    int c  = kg & 63;                        // channel
    int o  = nt * 16 + n;                    // output channel
    wp[idx] = (_Float16)cw[(o * 64 + c) * 9 + kt];
}

// ---------------------------------------------------------------------------
// Kernel A: 3x3 offset conv (only 9 of 18 channels are used downstream),
// BN + tanh + cumsum + off_new, one thread per pixel, weights in LDS.
// ---------------------------------------------------------------------------
__global__ __launch_bounds__(256) void kOffsets(const float* __restrict__ f,
                                                const float* __restrict__ ow,
                                                const float* __restrict__ ob,
                                                const float* __restrict__ bg,
                                                const float* __restrict__ bb,
                                                const float* __restrict__ bm,
                                                const float* __restrict__ bv,
                                                float* __restrict__ offws)
{
    __shared__ float ldsW[9 * 576];          // first 9 output channels of offset_w
    const int tid = threadIdx.x;
    for (int i = tid; i < 9 * 576; i += 256) ldsW[i] = ow[i];
    __syncthreads();

    const int pix = blockIdx.x * 256 + tid;  // 65536 pixels total
    const int b = pix >> 14;
    const int p = (pix >> 7) & 127;          // H-axis row
    const int q = pix & 127;                 // W-axis col

    float acc[9];
#pragma unroll
    for (int t = 0; t < 9; ++t) acc[t] = 0.0f;

    const float* fb = f + ((size_t)b << 20); // b * 64 * 16384
    for (int c = 0; c < 64; ++c) {
        const float* fc = fb + (c << 14);
#pragma unroll
        for (int dy = 0; dy < 3; ++dy) {
            int row = p + dy - 1;
            if ((unsigned)row >= 128u) continue;
#pragma unroll
            for (int dx = 0; dx < 3; ++dx) {
                int col = q + dx - 1;
                if ((unsigned)col >= 128u) continue;
                float fv = fc[(row << 7) + col];
                const float* wr = &ldsW[c * 9 + dy * 3 + dx];
#pragma unroll
                for (int t = 0; t < 9; ++t) acc[t] += fv * wr[t * 576];
            }
        }
    }

    float y[9], cs[9];
    float run = 0.0f;
#pragma unroll
    for (int t = 0; t < 9; ++t) {
        float v = acc[t] + ob[t];
        v = (v - bm[t]) * rsqrtf(bv[t] + 1e-5f);
        v = v * bg[t] + bb[t];
        v = tanhf(v);
        y[t] = v;
        run += v;
        cs[t] = run;
    }
#pragma unroll
    for (int t = 0; t < 9; ++t) {
        float on;
        if (t > 4)       on = cs[t] - cs[4];
        else if (t < 4)  on = cs[3] - cs[t] + y[t];
        else             on = 0.0f;
        offws[((b * 9 + t) << 14) + (p << 7) + q] = on;
    }
}

// ---------------------------------------------------------------------------
// Kernel B: bilinear gather -> LDS A-fragments (f16, WMMA swizzle) ->
// v_wmma_f32_16x16x32_f16 GEMM (M=32/block, N=64, K=576) -> bias, store x,
// hierarchical GroupNorm sum/sumsq reduction (LDS f32 atomics -> global).
// ---------------------------------------------------------------------------
__global__ __launch_bounds__(256) void kMain(const float* __restrict__ f,
                                             const float* __restrict__ offws,
                                             const _Float16* __restrict__ wpack,
                                             const float* __restrict__ cb,
                                             float* __restrict__ xws,
                                             float* __restrict__ gstats)
{
    __shared__ __align__(32) _Float16 ldsA[32 * 18 * 32]; // 32 rows x 576 K, fragment order
    __shared__ int   sx0[288], sx1[288], sy0[288], sy1[288];
    __shared__ float sw00[288], sw01[288], sw10[288], sw11[288];
    __shared__ float sStats[32];

    const int tid = threadIdx.x;
    const int m0  = blockIdx.x * 32;         // 32 GEMM rows per block (same b)
    const int b   = m0 >> 14;

    if (tid < 32) sStats[tid] = 0.0f;

    // Phase 0: per (row, tap) bilinear corner indices + weights (reference semantics,
    // incl. the exact-integer x-coordinate j+k-4 and clip-then-weight behavior).
    for (int p = tid; p < 288; p += 256) {
        int ml = p / 9, k = p - ml * 9;
        int m  = m0 + ml;
        int i  = (m >> 7) & 127;             // W-axis output row
        int j  = m & 127;                    // H-axis output col
        float yf = (float)i + offws[((b * 9 + k) << 14) + (i << 7) + j];
        int   xi = j + k - 4;                // integer x coordinate
        int   x0 = min(max(xi, 0), 127);
        int   x1 = min(max(xi + 1, 0), 127);
        float xf  = (float)xi;
        float wx0 = (float)x1 - xf;
        float wx1 = xf - (float)x0;
        int   fl = (int)floorf(yf);
        int   y0 = min(max(fl, 0), 127);
        int   y1 = min(max(fl + 1, 0), 127);
        float wy0 = (float)y1 - yf;
        float wy1 = yf - (float)y0;
        sx0[p] = x0 << 7; sx1[p] = x1 << 7; sy0[p] = y0; sy1[p] = y1;
        sw00[p] = wy0 * wx0; sw01[p] = wy0 * wx1;
        sw10[p] = wy1 * wx0; sw11[p] = wy1 * wx1;
    }
    __syncthreads();

    // Phase 1: gather 32 rows x 9 taps x 64 channels, write f16 into the
    // A-fragment order (contiguous: lane half h holds k = 16h + e of each 32-k step).
    for (int it = 0; it < 72; ++it) {
        int idx = it * 256 + tid;            // 18432 samples
        int c   = idx & 63;
        int p   = idx >> 6;
        int ml  = p / 9, k = p - ml * 9;
        const float* fc = f + ((size_t)(b * 64 + c) << 14);
        float v = fc[sx0[p] + sy0[p]] * sw00[p]
                + fc[sx0[p] + sy1[p]] * sw01[p]
                + fc[sx1[p] + sy0[p]] * sw10[p]
                + fc[sx1[p] + sy1[p]] * sw11[p];
        int kg   = k * 64 + c;               // K-dim index (matches kPack)
        int s    = kg >> 5;
        int kk   = kg & 31;
        int half = kk >> 4;
        int e    = kk & 15;
        ldsA[(ml * 18 + s) * 32 + half * 16 + e] = (_Float16)v;
    }
    __syncthreads();

    // Phase 2: 8 waves = 2 M-tiles x 4 N-tiles, 18 WMMAs each over K=576.
    const int wv   = tid >> 5;
    const int lane = tid & 31;
    const int mt   = wv >> 2;
    const int nt   = wv & 3;
    const int lrow = mt * 16 + (lane & 15);
    const int half = (lane >= 16) ? 1 : 0;
    const v16h* la = (const v16h*)ldsA;
    const v16h* wp = (const v16h*)wpack;

    v8f acc = {};
    for (int s = 0; s < 18; ++s) {
        v16h a  = la[(lrow * 18 + s) * 2 + half];
        v16h bm = wp[((s * 4 + nt) * 2 + half) * 16 + (lane & 15)];
        acc = __builtin_amdgcn_wmma_f32_16x16x32_f16(
            /*neg_a=*/false, a, /*neg_b=*/false, bm,
            /*c_mod=*/(short)0, acc, /*reuse_a=*/false, /*reuse_b=*/false);
    }

    // Epilogue: bias, store pre-GN x (NCHW), reduce group sums.
    const int   o    = nt * 16 + (lane & 15);
    const float bias = cb[o];
    float lsum = 0.0f, lsq = 0.0f;
    float* xo = xws + ((size_t)(b * 64 + o) << 14);
#pragma unroll
    for (int r = 0; r < 8; ++r) {
        int   mrow = mt * 16 + r + (half ? 8 : 0);   // D layout: lanes>=16 -> M+8
        float val  = acc[r] + bias;
        xo[(m0 + mrow) & 16383] = val;
        lsum += val;
        lsq  += val * val;
    }
    atomicAdd(&sStats[(o >> 2) * 2],     lsum);
    atomicAdd(&sStats[(o >> 2) * 2 + 1], lsq);
    __syncthreads();
    if (tid < 16) {
        atomicAdd(&gstats[(b * 16 + tid) * 2],     sStats[tid * 2]);
        atomicAdd(&gstats[(b * 16 + tid) * 2 + 1], sStats[tid * 2 + 1]);
    }
}

// ---------------------------------------------------------------------------
// Kernel C: GroupNorm finalize (population variance over 4ch*128*128) + ReLU.
// ---------------------------------------------------------------------------
__global__ __launch_bounds__(256) void kGnRelu(const float* __restrict__ xws,
                                               const float* __restrict__ gstats,
                                               const float* __restrict__ gg,
                                               const float* __restrict__ gb,
                                               float* __restrict__ out)
{
    int idx = blockIdx.x * 256 + threadIdx.x;      // 4194304 elements
    int o   = (idx >> 14) & 63;
    int b   = idx >> 20;
    int key = b * 16 + (o >> 2);
    const float inv = 1.0f / 65536.0f;             // 4 channels * 16384 px
    float mean = gstats[key * 2] * inv;
    float var  = gstats[key * 2 + 1] * inv - mean * mean;
    float xn   = (xws[idx] - mean) * rsqrtf(var + 1e-5f);
    float r    = xn * gg[o] + gb[o];
    out[idx]   = fmaxf(r, 0.0f);
}

// ---------------------------------------------------------------------------
extern "C" void kernel_launch(void* const* d_in, const int* in_sizes, int n_in,
                              void* d_out, int out_size, void* d_ws, size_t ws_size,
                              hipStream_t stream)
{
    const float* f        = (const float*)d_in[0];
    const float* offset_w = (const float*)d_in[1];
    const float* offset_b = (const float*)d_in[2];
    const float* bn_gamma = (const float*)d_in[3];
    const float* bn_beta  = (const float*)d_in[4];
    const float* bn_mean  = (const float*)d_in[5];
    const float* bn_var   = (const float*)d_in[6];
    const float* conv_w   = (const float*)d_in[7];
    const float* conv_b   = (const float*)d_in[8];
    const float* gn_gamma = (const float*)d_in[9];
    const float* gn_beta  = (const float*)d_in[10];

    char* ws = (char*)d_ws;
    float*    offws  = (float*)(ws + 0);
    _Float16* wpack  = (_Float16*)(ws + 2359296);
    float*    gstats = (float*)(ws + 2433024);
    float*    xws    = (float*)(ws + 2433536);

    kPack   <<<144,   256, 0, stream>>>(conv_w, wpack, gstats);
    kOffsets<<<256,   256, 0, stream>>>(f, offset_w, offset_b, bn_gamma, bn_beta,
                                        bn_mean, bn_var, offws);
    kMain   <<<2048,  256, 0, stream>>>(f, offws, wpack, conv_b, xws, gstats);
    kGnRelu <<<16384, 256, 0, stream>>>(xws, gstats, gn_gamma, gn_beta, (float*)d_out);
}
